// LYT_19447611916714
// MI455X (gfx1250) — matrix operand order repaired and checked by reading, not compiled, not run on hardware
//
#include <hip/hip_runtime.h>
#include <hip/hip_bf16.h>

// ---------------------------------------------------------------------------
// SS2D (VMamba selective-scan 2D) for gfx1250.
//   GEMMs : bf16 WMMA (v_wmma_f32_16x16x32_bf16), f32 accumulate,
//           multi-N-tile per wave for A-fragment register reuse.
//   Scan  : 2-pass segmented linear scan (32 segs x 128 steps),
//           LDS broadcast of per-step B/C/dt, atomic f32 scatter fusing the
//           4-direction recombination + D*u skip connection.
// ---------------------------------------------------------------------------

typedef __bf16 bf16_t;
typedef __attribute__((ext_vector_type(8)))  __bf16 v8bf;
typedef __attribute__((ext_vector_type(16))) __bf16 v16bf;
typedef __attribute__((ext_vector_type(8)))  float  v8f;

#define BATCH   4
#define HH      64
#define WW      64
#define LTOT    4096            // HH*WW
#define DMODEL  192
#define DINNER  384
#define NST     16
#define KG      4
#define RNK     12
#define CTOT    44              // RNK + 2*NST
#define CPAD    48
#define NSEG    32
#define SEGLEN  128             // LTOT / NSEG
#define MROWS   (BATCH*LTOT)    // 16384

// ---- WMMA fragment loads ---------------------------------------------------
// A (MxK, 16-bit, 16x32): lanes 0-15 hold rows M=0..15 with K = {0..7,16..23},
// lanes 16-31 same rows with K = {8..15,24..31}.  Row-major K-contiguous A
// means per-lane fragment = two contiguous 16B chunks at +0 and +16 elements.
__device__ inline v16bf ld_fragA(const bf16_t* p) {
    v8bf lo = *(const v8bf*)(p);
    v8bf hi = *(const v8bf*)(p + 16);
    return __builtin_shufflevector(lo, hi, 0,1,2,3,4,5,6,7,8,9,10,11,12,13,14,15);
}
// B (KxN, 16-bit, 32x16): lane n(0..15) holds col n with K = {0..7,16..23}, lanes
// 16..31 hold K = {8..15,24..31}: two contiguous 16B chunks at +0 and +16 when the
// operand is stored as [N,K] row-major (exactly the JAX weight layout).
__device__ inline v16bf ld_fragB(const bf16_t* p) {
    v8bf lo = *(const v8bf*)(p);
    v8bf hi = *(const v8bf*)(p + 16);
    return __builtin_shufflevector(lo, hi, 0,1,2,3,4,5,6,7,8,9,10,11,12,13,14,15);
}

__device__ inline float silu(float x) { return x / (1.f + __expf(-x)); }

// scan-step index l of direction k  ->  spatial flat index h*W+w
__device__ inline int perm_pos(int l, int k) {
    int t = (k & 2) ? (LTOT - 1 - l) : l;
    if (k & 1) { int w = t >> 6, h = t & 63; return h * WW + w; }  // H=W=64
    return t;
}

// multi-tile WMMA inner loop: one A fragment reused across NTILES B tiles
template <int NTILES>
__device__ inline void wmma_ktiles(const bf16_t* arow, const bf16_t* const (&brow)[NTILES],
                                   int K, int half, v8f (&acc)[NTILES]) {
    for (int k0 = 0; k0 < K; k0 += 32) {
        v16bf af = ld_fragA(arow + k0 + half * 8);
        #pragma unroll
        for (int t = 0; t < NTILES; ++t) {
            v16bf bf = ld_fragB(brow[t] + k0 + half * 8);
            acc[t] = __builtin_amdgcn_wmma_f32_16x16x32_bf16(false, af, false, bf,
                                                             (short)0, acc[t], false, false);
        }
    }
}

// ---- elementwise helpers ---------------------------------------------------
__global__ void ss2d_cvt_bf16(const float* __restrict__ s, bf16_t* __restrict__ d, int n) {
    int i = blockIdx.x * blockDim.x + threadIdx.x;
    if (i < n) d[i] = (bf16_t)s[i];
}

__global__ void ss2d_zero(float* __restrict__ p, int n) {
    int i = blockIdx.x * blockDim.x + threadIdx.x;
    if (i < n) p[i] = 0.f;
}

// x_proj_weight (KG,44,384) -> bf16 padded (KG,48,384), rows 44..47 zero
__global__ void ss2d_pack_xproj(const float* __restrict__ w, bf16_t* __restrict__ d) {
    int i = blockIdx.x * blockDim.x + threadIdx.x;
    if (i >= KG * CPAD * DINNER) return;
    int dd = i % DINNER; int c = (i / DINNER) % CPAD; int k = i / (DINNER * CPAD);
    float v = (c < CTOT) ? w[((size_t)k * CTOT + c) * DINNER + dd] : 0.f;
    d[i] = (bf16_t)v;
}

// ---- GEMM 1: xz = x @ in_proj_w^T, split into xp (f32) and z = silu (f32) --
// 1 M-tile x 4 N-tiles per wave; 1024 * 12 = 12288 waves.
__global__ void ss2d_gemm_inproj(const bf16_t* __restrict__ A, const bf16_t* __restrict__ W,
                                 float* __restrict__ xp, float* __restrict__ z) {
    const int K = DMODEL, NGRP = (2 * DINNER) / 64;          // 192, 12
    int wave = (blockIdx.x * blockDim.x + threadIdx.x) >> 5;
    int lane = threadIdx.x & 31;
    int mt = wave / NGRP, g = wave % NGRP;
    int half = lane >> 4, idx = lane & 15;
    const bf16_t* arow = A + (size_t)(mt * 16 + idx) * K;
    const bf16_t* brow[4];
    #pragma unroll
    for (int t = 0; t < 4; ++t)
        brow[t] = W + (size_t)(g * 64 + t * 16 + idx) * K;
    v8f acc[4] = {};
    wmma_ktiles<4>(arow, brow, K, half, acc);
    int row0 = mt * 16 + half * 8;
    #pragma unroll
    for (int t = 0; t < 4; ++t) {
        int col = g * 64 + t * 16 + idx;
        #pragma unroll
        for (int j = 0; j < 8; ++j) {
            float v = acc[t][j];
            size_t r = (size_t)(row0 + j);
            if (col < DINNER) xp[r * DINNER + col] = v;
            else              z [r * DINNER + (col - DINNER)] = silu(v);
        }
    }
}

// ---- depthwise 3x3 conv (SAME, zero pad) + bias + SiLU, channels-last ------
__global__ void ss2d_dwconv(const float* __restrict__ xp, const float* __restrict__ cw,
                            const float* __restrict__ cb, bf16_t* __restrict__ xcb) {
    int idx = blockIdx.x * blockDim.x + threadIdx.x;
    if (idx >= BATCH * LTOT * DINNER) return;
    int c = idx % DINNER;
    int w = (idx / DINNER) % WW;
    int h = (idx / (DINNER * WW)) % HH;
    int b = idx / (DINNER * WW * HH);
    float acc = cb[c];
    #pragma unroll
    for (int dy = -1; dy <= 1; ++dy) {
        int hh = h + dy; if (hh < 0 || hh >= HH) continue;
        #pragma unroll
        for (int dx = -1; dx <= 1; ++dx) {
            int ww = w + dx; if (ww < 0 || ww >= WW) continue;
            acc += xp[(((size_t)b * HH + hh) * WW + ww) * DINNER + c]
                 * cw[c * 9 + (dy + 1) * 3 + (dx + 1)];
        }
    }
    xcb[idx] = (bf16_t)silu(acc);
}

// ---- GEMM 2: x_dbl[b,k,l,c] = sum_d xc[b,perm_k(l),d] * Wx[k,c,d] ----------
// All 3 N-tiles (48 padded cols) per wave: permuted A row loaded once.
// 16 * 256 = 4096 waves.
__global__ void ss2d_gemm_xdbl(const bf16_t* __restrict__ xc, const bf16_t* __restrict__ Wx,
                               float* __restrict__ xdbl) {
    const int K = DINNER, MT = LTOT / 16;                    // 384, 256
    int wave = (blockIdx.x * blockDim.x + threadIdx.x) >> 5;
    int lane = threadIdx.x & 31;
    int bk = wave / MT;
    int mt = wave % MT;
    int b = bk >> 2, k = bk & 3;
    int half = lane >> 4, idx = lane & 15;
    int l   = mt * 16 + idx;
    int pos = perm_pos(l, k);
    const bf16_t* arow = xc + ((size_t)b * LTOT + pos) * K;
    const bf16_t* brow[3];
    #pragma unroll
    for (int t = 0; t < 3; ++t)
        brow[t] = Wx + ((size_t)k * CPAD + t * 16 + idx) * K;
    v8f acc[3] = {};
    wmma_ktiles<3>(arow, brow, K, half, acc);
    int row0 = mt * 16 + half * 8;
    #pragma unroll
    for (int t = 0; t < 3; ++t) {
        int col = t * 16 + idx;
        if (col < CTOT) {
            #pragma unroll
            for (int j = 0; j < 8; ++j)
                xdbl[(((size_t)bk * LTOT) + row0 + j) * CTOT + col] = acc[t][j];
        }
    }
}

// ---- segmented selective scan ----------------------------------------------
// pass A (EMIT=false): per segment, h from 0 + running prod(dA)  -> hseg,aprod
// pass C (EMIT=true) : h from hinit, emit y + D*u via atomic scatter
template <bool EMIT>
__global__ void ss2d_scan_pass(const float* __restrict__ xdbl, const bf16_t* __restrict__ xc,
                               const float* __restrict__ Alogs, const float* __restrict__ dtw,
                               const float* __restrict__ dtb, const float* __restrict__ Dsv,
                               const float* __restrict__ hinit,
                               float* __restrict__ hseg, float* __restrict__ aprod,
                               float* __restrict__ ysum) {
    int seg = blockIdx.x;            // 0..31
    int bk  = blockIdx.y;            // 0..15
    int b = bk >> 2, k = bk & 3;
    int d = threadIdx.x;             // 0..383
    int ch = k * DINNER + d;

    float A[NST];
    #pragma unroll
    for (int n = 0; n < NST; ++n) A[n] = -__expf(Alogs[(size_t)ch * NST + n]);
    float wr[RNK];
    #pragma unroll
    for (int r = 0; r < RNK; ++r) wr[r] = dtw[(size_t)ch * RNK + r];
    float bias = dtb[ch];
    float Dv = EMIT ? Dsv[ch] : 0.f;

    size_t hbase = (((size_t)bk * NSEG + seg) * DINNER + d) * NST;
    float h[NST], ap[NST];
    #pragma unroll
    for (int n = 0; n < NST; ++n) { h[n] = EMIT ? hinit[hbase + n] : 0.f; ap[n] = 1.f; }

    __shared__ float sBC[CTOT];
    const float*  xd  = xdbl + (size_t)bk * LTOT * CTOT;
    const bf16_t* xcb = xc   + (size_t)b  * LTOT * DINNER;
    int l0 = seg * SEGLEN;

    for (int i = 0; i < SEGLEN; ++i) {
        int l = l0 + i;
        __syncthreads();
        if (d < CTOT) sBC[d] = xd[(size_t)l * CTOT + d];
        __syncthreads();

        float dtr = bias;
        #pragma unroll
        for (int r = 0; r < RNK; ++r) dtr += wr[r] * sBC[r];
        float delta = (dtr > 20.f) ? dtr : log1pf(__expf(dtr));   // softplus

        int pos = perm_pos(l, k);
        float u  = (float)xcb[(size_t)pos * DINNER + d];
        float du = delta * u;
        float y  = Dv * u;
        #pragma unroll
        for (int n = 0; n < NST; ++n) {
            float a = __expf(delta * A[n]);
            h[n] = a * h[n] + du * sBC[RNK + n];
            if (EMIT) y += h[n] * sBC[RNK + NST + n];
            else      ap[n] *= a;
        }
        if (EMIT) atomicAdd(&ysum[((size_t)b * LTOT + pos) * DINNER + d], y);
    }
    if (!EMIT) {
        #pragma unroll
        for (int n = 0; n < NST; ++n) { hseg[hbase + n] = h[n]; aprod[hbase + n] = ap[n]; }
    }
}

// ---- pass B: chain segment boundary states (tiny, fully parallel over chan,n)
__global__ void ss2d_scan_fix(const float* __restrict__ hseg, const float* __restrict__ aprod,
                              float* __restrict__ hinit) {
    int idx = blockIdx.x * blockDim.x + threadIdx.x;    // (bk,d,n)
    if (idx >= BATCH * KG * DINNER * NST) return;
    int n  = idx & (NST - 1);
    int d  = (idx >> 4) % DINNER;
    int bk = idx / (NST * DINNER);
    float h = 0.f;
    for (int s = 0; s < NSEG; ++s) {
        size_t base = (((size_t)bk * NSEG + s) * DINNER + d) * NST + n;
        hinit[base] = h;
        h = aprod[base] * h + hseg[base];
    }
}

// ---- LayerNorm(y) * z, emit bf16 -------------------------------------------
__global__ void ss2d_ln_mul(const float* __restrict__ ysum, const float* __restrict__ z,
                            const float* __restrict__ gamma, const float* __restrict__ beta,
                            bf16_t* __restrict__ yz) {
    int row = blockIdx.x;            // 0..16383
    int t = threadIdx.x;             // 0..127
    const float* yr = ysum + (size_t)row * DINNER;
    float v0 = yr[t], v1 = yr[t + 128], v2 = yr[t + 256];
    __shared__ float red[128];
    red[t] = v0 + v1 + v2;
    __syncthreads();
    for (int o = 64; o > 0; o >>= 1) { if (t < o) red[t] += red[t + o]; __syncthreads(); }
    float mu = red[0] * (1.f / DINNER);
    __syncthreads();
    float d0 = v0 - mu, d1 = v1 - mu, d2 = v2 - mu;
    red[t] = d0 * d0 + d1 * d1 + d2 * d2;
    __syncthreads();
    for (int o = 64; o > 0; o >>= 1) { if (t < o) red[t] += red[t + o]; __syncthreads(); }
    float inv = rsqrtf(red[0] * (1.f / DINNER) + 1e-5f);
    const float* zr = z + (size_t)row * DINNER;
    #pragma unroll
    for (int i = 0; i < 3; ++i) {
        int c = t + i * 128;
        float val = ((i == 0 ? v0 : (i == 1 ? v1 : v2)) - mu) * inv * gamma[c] + beta[c];
        yz[(size_t)row * DINNER + c] = (bf16_t)(val * zr[c]);
    }
}

// ---- GEMM 3: out = yz @ out_proj_w^T ---------------------------------------
// 1 M-tile x 4 N-tiles per wave; 1024 * 3 = 3072 waves.
__global__ void ss2d_gemm_out(const bf16_t* __restrict__ A, const bf16_t* __restrict__ W,
                              float* __restrict__ out) {
    const int K = DINNER, NGRP = DMODEL / 64;            // 384, 3
    int wave = (blockIdx.x * blockDim.x + threadIdx.x) >> 5;
    int lane = threadIdx.x & 31;
    int mt = wave / NGRP, g = wave % NGRP;
    int half = lane >> 4, idx = lane & 15;
    const bf16_t* arow = A + (size_t)(mt * 16 + idx) * K;
    const bf16_t* brow[4];
    #pragma unroll
    for (int t = 0; t < 4; ++t)
        brow[t] = W + (size_t)(g * 64 + t * 16 + idx) * K;
    v8f acc[4] = {};
    wmma_ktiles<4>(arow, brow, K, half, acc);
    int row0 = mt * 16 + half * 8;
    #pragma unroll
    for (int t = 0; t < 4; ++t) {
        int col = g * 64 + t * 16 + idx;
        #pragma unroll
        for (int j = 0; j < 8; ++j)
            out[(size_t)(row0 + j) * DMODEL + col] = acc[t][j];
    }
}

// ---------------------------------------------------------------------------
extern "C" void kernel_launch(void* const* d_in, const int* in_sizes, int n_in,
                              void* d_out, int out_size, void* d_ws, size_t ws_size,
                              hipStream_t stream) {
    (void)in_sizes; (void)n_in; (void)out_size; (void)ws_size;
    const float* x     = (const float*)d_in[0];
    const float* ipw   = (const float*)d_in[1];
    const float* convw = (const float*)d_in[2];
    const float* convb = (const float*)d_in[3];
    const float* xpw   = (const float*)d_in[4];
    const float* dtw   = (const float*)d_in[5];
    const float* dtb   = (const float*)d_in[6];
    const float* alogs = (const float*)d_in[7];
    const float* Dsp   = (const float*)d_in[8];
    const float* gamma = (const float*)d_in[9];
    const float* beta  = (const float*)d_in[10];
    const float* opw   = (const float*)d_in[11];
    float* out = (float*)d_out;

    // workspace carve-out (~114 MiB), with reuse: xp->ysum, hseg->yz
    char* ws = (char*)d_ws;
    size_t off = 0;
    auto carve = [&](size_t bytes) { void* p = ws + off; off += (bytes + 255) & ~(size_t)255; return p; };
    float*  xp_ysum = (float*) carve((size_t)MROWS * DINNER * 4);          // 25.2 MB
    float*  zbuf    = (float*) carve((size_t)MROWS * DINNER * 4);          // 25.2 MB
    bf16_t* xcb     = (bf16_t*)carve((size_t)MROWS * DINNER * 2);          // 12.6 MB
    bf16_t* xbf     = (bf16_t*)carve((size_t)MROWS * DMODEL * 2);          //  6.3 MB
    bf16_t* wip     = (bf16_t*)carve((size_t)2 * DINNER * DMODEL * 2);     //  0.3 MB
    bf16_t* wxp     = (bf16_t*)carve((size_t)KG * CPAD * DINNER * 2);      //  0.15 MB
    bf16_t* wout    = (bf16_t*)carve((size_t)DMODEL * DINNER * 2);         //  0.15 MB
    float*  xdbl    = (float*) carve((size_t)BATCH * KG * LTOT * CTOT * 4);// 11.5 MB
    float*  hseg_yz = (float*) carve((size_t)16 * NSEG * DINNER * NST * 4);// 12.6 MB
    float*  aprod   = (float*) carve((size_t)16 * NSEG * DINNER * NST * 4);// 12.6 MB
    float*  hinit   = (float*) carve((size_t)16 * NSEG * DINNER * NST * 4);// 12.6 MB
    bf16_t* yz = (bf16_t*)hseg_yz;

    // 1) pack operands to bf16
    ss2d_cvt_bf16<<<(MROWS * DMODEL + 255) / 256, 256, 0, stream>>>(x, xbf, MROWS * DMODEL);
    ss2d_cvt_bf16<<<(2 * DINNER * DMODEL + 255) / 256, 256, 0, stream>>>(ipw, wip, 2 * DINNER * DMODEL);
    ss2d_cvt_bf16<<<(DMODEL * DINNER + 255) / 256, 256, 0, stream>>>(opw, wout, DMODEL * DINNER);
    ss2d_pack_xproj<<<(KG * CPAD * DINNER + 255) / 256, 256, 0, stream>>>(xpw, wxp);

    // 2) in_proj (WMMA, 4 N-tiles/wave): 12288 waves
    ss2d_gemm_inproj<<<1536, 256, 0, stream>>>(xbf, wip, xp_ysum, zbuf);

    // 3) depthwise conv + SiLU -> bf16 channels-last
    ss2d_dwconv<<<(MROWS * DINNER + 255) / 256, 256, 0, stream>>>(xp_ysum, convw, convb, xcb);

    // 4) x_proj per direction (WMMA, permuted A rows, 3 N-tiles/wave): 4096 waves
    ss2d_gemm_xdbl<<<512, 256, 0, stream>>>(xcb, wxp, xdbl);

    // 5) segmented scan: pass A, fix-up, zero accumulator, pass C (emit)
    ss2d_scan_pass<false><<<dim3(NSEG, 16), DINNER, 0, stream>>>(
        xdbl, xcb, alogs, dtw, dtb, Dsp, hinit, hseg_yz, aprod, nullptr);
    ss2d_scan_fix<<<(BATCH * KG * DINNER * NST + 255) / 256, 256, 0, stream>>>(hseg_yz, aprod, hinit);
    ss2d_zero<<<(MROWS * DINNER + 255) / 256, 256, 0, stream>>>(xp_ysum, MROWS * DINNER);
    ss2d_scan_pass<true><<<dim3(NSEG, 16), DINNER, 0, stream>>>(
        xdbl, xcb, alogs, dtw, dtb, Dsp, hinit, nullptr, nullptr, xp_ysum);

    // 6) LayerNorm * z -> bf16
    ss2d_ln_mul<<<MROWS, 128, 0, stream>>>(xp_ysum, zbuf, gamma, beta, yz);

    // 7) out_proj (WMMA, 4 N-tiles/wave): 3072 waves
    ss2d_gemm_out<<<384, 256, 0, stream>>>(yz, wout, out);
}
